// Decoder_37391985279218
// MI455X (gfx1250) — matrix operand order repaired and checked by reading, not compiled, not run on hardware
//
#include <hip/hip_runtime.h>
#include <hip/hip_bf16.h>

// ---------------------------------------------------------------------------
// MI455X (gfx1250) decoder forward.
// fp32 weights stream once from HBM (~1.9GB -> 82us floor @ 23.3TB/s),
// converted to bf16 in-register (v_cvt_pk_bf16_f32); all matmuls use
// v_wmma_f32_16x16x32_bf16. GEMM is double-buffered in LDS; bf16 operands
// staged with gfx1250 async global->LDS loads (ASYNCcnt).
// ---------------------------------------------------------------------------

typedef __attribute__((ext_vector_type(16))) __bf16 v16bf;
typedef __attribute__((ext_vector_type(4)))  __bf16 v4bf;
typedef __attribute__((ext_vector_type(8)))  float v8f;

#define S_TOK 1024
#define D_MODEL 4096
#define N_HEADS 32
#define D_HEAD 128
#define F_FFN 8192
#define M_MEM 512
#define KV_LEN 1536
#define V_VOCAB 8192
#define N_LAYERS 2

#if defined(__AMDGCN__) && __has_builtin(__builtin_amdgcn_global_load_async_to_lds_b128)
#define HAVE_ASYNC 1
#else
#define HAVE_ASYNC 0
#endif

__device__ __forceinline__ unsigned short f2bf(float f) {
  return __builtin_bit_cast(unsigned short, static_cast<__bf16>(f));
}

union BF16x16 { v16bf v; uint4 u[2]; };

#if HAVE_ASYNC
// builtin signature (probed via hipcc diagnostic): param0 = addrspace(1)
// "int __vector(4)"* (global src), param1 = addrspace(3) dst, imm offset, imm cpol.
typedef int v4i_raw __attribute__((vector_size(16)));
typedef __attribute__((address_space(1))) v4i_raw as1_v4i;
typedef __attribute__((address_space(3))) v4i_raw as3_v4i;
__device__ __forceinline__ void async_cp16(const void* g, void* l) {
  __builtin_amdgcn_global_load_async_to_lds_b128(
      (as1_v4i*)(unsigned long long)g,
      (as3_v4i*)(unsigned)(unsigned long long)l, 0, 0);
}
#endif

template <int N> __device__ __forceinline__ void wait_async() {
#if HAVE_ASYNC
#if __has_builtin(__builtin_amdgcn_s_wait_asynccnt)
  __builtin_amdgcn_s_wait_asynccnt(N);
#else
  asm volatile("s_wait_asynccnt %0" ::"i"(N) : "memory");
#endif
#endif
}

// ---------------------------------------------------------------------------
// Generic WMMA GEMM:  C[b] = A[b](bf16,[M,K]) * B[b]([N,K])^T   (row-major)
// TB=float : weight operand, fp32->bf16 converted while staging
// TB=ushort: bf16 operand (attention), async-copied straight to LDS
// MODE 0: C fp32          MODE 1: C fp32 = acc + R (fused residual)
// MODE 2: C bf16(ushort)  MODE 3: C fp32 = causal_mask(scale*acc)
// Tile 128x128x64, 256 thr = 8 waves, wave -> 32x64 (2x4 wmma tiles),
// double-buffered LDS software pipeline.
// ---------------------------------------------------------------------------
#define BM 128
#define BN 128
#define BK 64
#define LDSS 72   // ushort stride: 144B rows (16B aligned, bank padding)

template <typename TB, int MODE>
__global__ __launch_bounds__(256)
void gemm_wmma(const unsigned short* __restrict__ A, const TB* __restrict__ B,
               void* __restrict__ Cout, const float* __restrict__ R,
               int Md, int Nd, int Kd,
               long long sA, long long sB, long long sC, int ldc, float scale) {
  constexpr bool BF16B = (sizeof(TB) == 2);
  __shared__ unsigned short As[2][BM * LDSS];
  __shared__ unsigned short Bs[2][BN * LDSS];

  const int bz = blockIdx.z;
  A += (long long)bz * sA;
  B += (long long)bz * sB;

  const int tid = threadIdx.x;
  const int lane = tid & 31;
  const int wv = tid >> 5;
  const int waveM = wv >> 1;        // 0..3
  const int waveN = wv & 1;         // 0..1
  const int lane16 = lane & 15;
  const int laneHalf = lane >> 4;
  const int rowBlk = blockIdx.y * BM;
  const int colBlk = blockIdx.x * BN;

  // staging chunk coords: A/bf16-B = 4 x b128 per thread, fp32-B = 8 x b128
  int arow[4], acol[4];
#pragma unroll
  for (int j = 0; j < 4; ++j) {
    int c = tid + j * 256;            // 0..1023
    arow[j] = c >> 3; acol[j] = (c & 7) * 8;
  }
  int brow[8], bcol[8];
#pragma unroll
  for (int j = 0; j < 8; ++j) {
    int c = tid + j * 256;            // 0..2047
    brow[j] = c >> 4; bcol[j] = (c & 15) * 4;
  }

  uint4  aReg[4];
  uint4  bRegH[4];
  float4 bRegF[8];

  auto issueNext = [&](int buf, int k0) {
#if HAVE_ASYNC
#pragma unroll
    for (int j = 0; j < 4; ++j)
      async_cp16(A + (size_t)(rowBlk + arow[j]) * Kd + k0 + acol[j],
                 &As[buf][arow[j] * LDSS + acol[j]]);
#else
#pragma unroll
    for (int j = 0; j < 4; ++j)
      aReg[j] = *(const uint4*)(A + (size_t)(rowBlk + arow[j]) * Kd + k0 + acol[j]);
#endif
    if constexpr (BF16B) {
#if HAVE_ASYNC
#pragma unroll
      for (int j = 0; j < 4; ++j)
        async_cp16((const unsigned short*)B +
                       (size_t)(colBlk + arow[j]) * Kd + k0 + acol[j],
                   &Bs[buf][arow[j] * LDSS + acol[j]]);
#else
#pragma unroll
      for (int j = 0; j < 4; ++j)
        bRegH[j] = *(const uint4*)((const unsigned short*)B +
                                   (size_t)(colBlk + arow[j]) * Kd + k0 + acol[j]);
#endif
    } else {
#pragma unroll
      for (int j = 0; j < 8; ++j)
        bRegF[j] = *(const float4*)((const float*)B +
                                    (size_t)(colBlk + brow[j]) * Kd + k0 + bcol[j]);
      if (k0 + 2 * BK < Kd)  // warm L2 two tiles ahead (global_prefetch_b8)
        __builtin_prefetch((const char*)((const float*)B +
                                         (size_t)colBlk * Kd + k0 + 2 * BK), 0, 1);
    }
  };

  auto storeNext = [&](int buf) {
#if !HAVE_ASYNC
#pragma unroll
    for (int j = 0; j < 4; ++j)
      *(uint4*)&As[buf][arow[j] * LDSS + acol[j]] = aReg[j];
    if constexpr (BF16B) {
#pragma unroll
      for (int j = 0; j < 4; ++j)
        *(uint4*)&Bs[buf][arow[j] * LDSS + acol[j]] = bRegH[j];
    }
#endif
    if constexpr (!BF16B) {
#pragma unroll
      for (int j = 0; j < 8; ++j) {
        // native bf16 vector build -> 2x v_cvt_pk_bf16_f32 + ds_store_b64
        v4bf p = { (__bf16)bRegF[j].x, (__bf16)bRegF[j].y,
                   (__bf16)bRegF[j].z, (__bf16)bRegF[j].w };
        *(v4bf*)&Bs[buf][brow[j] * LDSS + bcol[j]] = p;
      }
    }
  };

  v8f acc[2][4];
#pragma unroll
  for (int a = 0; a < 2; ++a)
#pragma unroll
    for (int b = 0; b < 4; ++b)
      acc[a][b] = (v8f){0.f, 0.f, 0.f, 0.f, 0.f, 0.f, 0.f, 0.f};

  const int nk = Kd / BK;
  // prologue: stage tile 0
  issueNext(0, 0);
  storeNext(0);

  for (int i = 0; i < nk; ++i) {
    const int buf = i & 1;
    const bool hasNext = (i + 1 < nk);
    if (hasNext) issueNext(buf ^ 1, (i + 1) * BK);   // overlap with compute
    if (hasNext) wait_async<(BF16B ? 8 : 4)>();      // tile i done, i+1 in flight
    else         wait_async<0>();
    __syncthreads();                                  // tile i visible to all waves

#pragma unroll
    for (int kk = 0; kk < 2; ++kk) {                  // two 16x16x32 K-steps
      v16bf afrag[2], bfrag[4];
#pragma unroll
      for (int tm = 0; tm < 2; ++tm) {
        const unsigned short* rb =
            &As[buf][(waveM * 32 + tm * 16 + lane16) * LDSS + kk * 32];
        BF16x16 t;
        t.u[0] = *(const uint4*)(rb + laneHalf * 8);
        t.u[1] = *(const uint4*)(rb + 16 + laneHalf * 8);
        afrag[tm] = t.v;
      }
#pragma unroll
      for (int tn = 0; tn < 4; ++tn) {
        const unsigned short* cb =
            &Bs[buf][(waveN * 64 + tn * 16 + lane16) * LDSS + kk * 32 +
                     laneHalf * 8];
        BF16x16 t;
        t.u[0] = *(const uint4*)(cb);
        t.u[1] = *(const uint4*)(cb + 16);
        bfrag[tn] = t.v;
      }
#pragma unroll
      for (int tm = 0; tm < 2; ++tm)
#pragma unroll
        for (int tn = 0; tn < 4; ++tn)
          acc[tm][tn] = __builtin_amdgcn_wmma_f32_16x16x32_bf16(
              false, afrag[tm], false, bfrag[tn], (short)0, acc[tm][tn],
              false, false);
    }

    if (hasNext) storeNext(buf ^ 1);                  // convert/land next B tile
    __syncthreads();                                  // readers done before reuse
  }

  // epilogue (C layout: lane=col, vgpr i -> row i + 8*laneHalf)
#pragma unroll
  for (int tm = 0; tm < 2; ++tm)
#pragma unroll
    for (int tn = 0; tn < 4; ++tn) {
      int col = colBlk + waveN * 64 + tn * 16 + lane16;
#pragma unroll
      for (int i = 0; i < 8; ++i) {
        int row = rowBlk + waveM * 32 + tm * 16 + i + 8 * laneHalf;
        float c = acc[tm][tn][i];
        size_t idx = (size_t)row * ldc + col + (size_t)bz * sC;
        if constexpr (MODE == 0) {
          ((float*)Cout)[idx] = c;
        } else if constexpr (MODE == 1) {
          ((float*)Cout)[idx] = c + R[idx];
        } else if constexpr (MODE == 2) {
          ((unsigned short*)Cout)[idx] = f2bf(c);
        } else {  // causal scores: qpos = 512+row, kpos = col
          c *= scale;
          if (col > 512 + row) c = -1e30f;
          ((float*)Cout)[idx] = c;
        }
      }
    }
}

// ---------------------------------------------------------------------------
// elementwise kernels
// ---------------------------------------------------------------------------
__global__ __launch_bounds__(256)
void gather_embed_kernel(const int* __restrict__ ids,
                         const float* __restrict__ embed,
                         float* __restrict__ h) {
  int d = blockIdx.x * 256 + threadIdx.x;
  int s = blockIdx.y;
  h[(size_t)s * D_MODEL + d] = embed[(size_t)ids[s] * D_MODEL + d];
}

__global__ __launch_bounds__(256)
void rmsnorm_kernel(const float* __restrict__ x, const float* __restrict__ w,
                    unsigned short* __restrict__ out) {
  __shared__ float red[8];
  const int row = blockIdx.x;
  const float* xr = x + (size_t)row * D_MODEL;
  float ss = 0.f;
  for (int i = threadIdx.x; i < D_MODEL; i += 256) { float v = xr[i]; ss += v * v; }
#pragma unroll
  for (int off = 16; off > 0; off >>= 1) ss += __shfl_xor(ss, off, 32);
  if ((threadIdx.x & 31) == 0) red[threadIdx.x >> 5] = ss;
  __syncthreads();
  float tot = 0.f;
#pragma unroll
  for (int i = 0; i < 8; ++i) tot += red[i];
  float r = rsqrtf(tot / (float)D_MODEL + 1e-5f);
  for (int i = threadIdx.x; i < D_MODEL; i += 256)
    out[(size_t)row * D_MODEL + i] = f2bf(xr[i] * r * w[i]);
}

__global__ __launch_bounds__(256)
void f32_to_bf16_kernel(const float* __restrict__ in,
                        unsigned short* __restrict__ out, int n) {
  int i = blockIdx.x * 256 + threadIdx.x;
  if (i < n) out[i] = f2bf(in[i]);
}

__global__ __launch_bounds__(256)
void rope_q_kernel(const float* __restrict__ q, unsigned short* __restrict__ Qr) {
  int t = blockIdx.x * 256 + threadIdx.x;     // H*S*64
  int i = t & 63;
  int s = (t >> 6) & (S_TOK - 1);
  int hh = t >> 16;
  float x0 = q[(size_t)s * D_MODEL + hh * D_HEAD + i];
  float x1 = q[(size_t)s * D_MODEL + hh * D_HEAD + i + 64];
  float inv = __powf(10000.f, -(float)(2 * i) / 128.f);
  float ang = (float)(512 + s) * inv;
  float cs = __cosf(ang), sn = __sinf(ang);
  size_t o = (size_t)hh * S_TOK * D_HEAD + (size_t)s * D_HEAD;
  Qr[o + i]      = f2bf(x0 * cs - x1 * sn);
  Qr[o + i + 64] = f2bf(x1 * cs + x0 * sn);
}

__global__ __launch_bounds__(256)
void rope_k_kernel(const float* __restrict__ mk, const float* __restrict__ k,
                   unsigned short* __restrict__ Kr) {
  int t = blockIdx.x * 256 + threadIdx.x;     // H*1536*64
  int i = t & 63;
  int p = (t >> 6) % KV_LEN;
  int hh = t / (64 * KV_LEN);
  const float* src = (p < M_MEM) ? (mk + (size_t)p * D_MODEL)
                                 : (k + (size_t)(p - M_MEM) * D_MODEL);
  float x0 = src[hh * D_HEAD + i];
  float x1 = src[hh * D_HEAD + i + 64];
  float inv = __powf(10000.f, -(float)(2 * i) / 128.f);
  float ang = (float)p * inv;
  float cs = __cosf(ang), sn = __sinf(ang);
  size_t o = (size_t)hh * KV_LEN * D_HEAD + (size_t)p * D_HEAD;
  Kr[o + i]      = f2bf(x0 * cs - x1 * sn);
  Kr[o + i + 64] = f2bf(x1 * cs + x0 * sn);
}

__global__ __launch_bounds__(256)
void vt_kernel(const float* __restrict__ mv, const float* __restrict__ v,
               unsigned short* __restrict__ Vt) {
  int t = blockIdx.x * 256 + threadIdx.x;     // H*1536*128
  int d = t & 127;
  int p = (t >> 7) % KV_LEN;
  int hh = t / (128 * KV_LEN);
  float val = (p < M_MEM) ? mv[(size_t)p * D_MODEL + hh * D_HEAD + d]
                          : v[(size_t)(p - M_MEM) * D_MODEL + hh * D_HEAD + d];
  Vt[(size_t)hh * D_HEAD * KV_LEN + (size_t)d * KV_LEN + p] = f2bf(val);
}

__global__ __launch_bounds__(256)
void softmax_kernel(const float* __restrict__ sc, unsigned short* __restrict__ P) {
  __shared__ float redA[8], redB[8];
  const size_t row = blockIdx.x;              // H*S rows, length 1536
  const float* s = sc + row * KV_LEN;
  float mx = -3.0e38f;
  for (int i = threadIdx.x; i < KV_LEN; i += 256) mx = fmaxf(mx, s[i]);
#pragma unroll
  for (int off = 16; off > 0; off >>= 1) mx = fmaxf(mx, __shfl_xor(mx, off, 32));
  if ((threadIdx.x & 31) == 0) redA[threadIdx.x >> 5] = mx;
  __syncthreads();
#pragma unroll
  for (int i = 0; i < 8; ++i) mx = fmaxf(mx, redA[i]);
  float sum = 0.f;
  for (int i = threadIdx.x; i < KV_LEN; i += 256) sum += __expf(s[i] - mx);
#pragma unroll
  for (int off = 16; off > 0; off >>= 1) sum += __shfl_xor(sum, off, 32);
  if ((threadIdx.x & 31) == 0) redB[threadIdx.x >> 5] = sum;
  __syncthreads();
  float tot = 0.f;
#pragma unroll
  for (int i = 0; i < 8; ++i) tot += redB[i];
  float invs = 1.f / tot;
  for (int i = threadIdx.x; i < KV_LEN; i += 256)
    P[row * KV_LEN + i] = f2bf(__expf(s[i] - mx) * invs);
}

__global__ __launch_bounds__(256)
void silu_mul_kernel(const float* __restrict__ g, const float* __restrict__ u,
                     unsigned short* __restrict__ ff, int n) {
  int i = blockIdx.x * 256 + threadIdx.x;
  if (i < n) {
    float gv = g[i];
    ff[i] = f2bf(gv / (1.f + __expf(-gv)) * u[i]);
  }
}

// ---------------------------------------------------------------------------
// host orchestration
// ---------------------------------------------------------------------------
extern "C" void kernel_launch(void* const* d_in, const int* in_sizes, int n_in,
                              void* d_out, int out_size, void* d_ws, size_t ws_size,
                              hipStream_t stream) {
  (void)in_sizes; (void)n_in; (void)out_size; (void)ws_size;
  const int*   ids    = (const int*)d_in[0];
  const float* memory = (const float*)d_in[1];
  const float* embed  = (const float*)d_in[2];
  const float* Wq     = (const float*)d_in[3];
  const float* Wk     = (const float*)d_in[4];
  const float* Wv     = (const float*)d_in[5];
  const float* Wo     = (const float*)d_in[6];
  const float* Wg     = (const float*)d_in[7];
  const float* Wu     = (const float*)d_in[8];
  const float* Wd     = (const float*)d_in[9];
  const float* Wmk    = (const float*)d_in[10];
  const float* Wmv    = (const float*)d_in[11];
  const float* ln1    = (const float*)d_in[12];
  const float* ln2    = (const float*)d_in[13];
  const float* normw  = (const float*)d_in[14];
  const float* lmh    = (const float*)d_in[15];

  const unsigned long long MB = 1024ull * 1024ull;
  unsigned char* ws = (unsigned char*)d_ws;
  float*          hB   = (float*)(ws + 0);                // 16MB
  unsigned short* xb   = (unsigned short*)(ws + 16*MB);   // 8MB
  unsigned short* memb = (unsigned short*)(ws + 24*MB);   // 4MB
  float*          qf   = (float*)(ws + 28*MB);            // 16MB
  float*          kf   = (float*)(ws + 44*MB);            // 16MB
  float*          vf   = (float*)(ws + 60*MB);            // 16MB
  float*          mkf  = (float*)(ws + 76*MB);            // 8MB
  float*          mvf  = (float*)(ws + 84*MB);            // 8MB
  unsigned short* Qr   = (unsigned short*)(ws + 92*MB);   // 8MB
  unsigned short* Kr   = (unsigned short*)(ws + 100*MB);  // 12MB
  unsigned short* Vt   = (unsigned short*)(ws + 112*MB);  // 12MB
  unsigned short* ob   = (unsigned short*)(ws + 124*MB);  // 8MB
  float*          scor = (float*)(ws + 128*MB);           // 192MB (then g/u)
  float*          gf   = (float*)(ws + 128*MB);           // 32MB
  float*          uf   = (float*)(ws + 160*MB);           // 32MB
  unsigned short* Pb   = (unsigned short*)(ws + 320*MB);  // 96MB (then ff)
  unsigned short* ffb  = (unsigned short*)(ws + 320*MB);  // 16MB

  const dim3 blk(256);
  const float scale = 0.08838834764831845f;  // 1/sqrt(128)

  gather_embed_kernel<<<dim3(D_MODEL / 256, S_TOK), blk, 0, stream>>>(ids, embed, hB);

  for (int l = 0; l < N_LAYERS; ++l) {
    const float* Wq_l  = Wq  + (size_t)l * D_MODEL * D_MODEL;
    const float* Wk_l  = Wk  + (size_t)l * D_MODEL * D_MODEL;
    const float* Wv_l  = Wv  + (size_t)l * D_MODEL * D_MODEL;
    const float* Wo_l  = Wo  + (size_t)l * D_MODEL * D_MODEL;
    const float* Wmk_l = Wmk + (size_t)l * D_MODEL * D_MODEL;
    const float* Wmv_l = Wmv + (size_t)l * D_MODEL * D_MODEL;
    const float* Wg_l  = Wg  + (size_t)l * F_FFN * D_MODEL;
    const float* Wu_l  = Wu  + (size_t)l * F_FFN * D_MODEL;
    const float* Wd_l  = Wd  + (size_t)l * D_MODEL * F_FFN;

    rmsnorm_kernel<<<S_TOK, blk, 0, stream>>>(hB, ln1 + l * D_MODEL, xb);
    f32_to_bf16_kernel<<<(M_MEM * D_MODEL) / 256, blk, 0, stream>>>(
        memory + (size_t)l * M_MEM * D_MODEL, memb, M_MEM * D_MODEL);

    gemm_wmma<float, 0><<<dim3(32, 8, 1), blk, 0, stream>>>(
        xb, Wq_l, qf, nullptr, S_TOK, D_MODEL, D_MODEL, 0, 0, 0, D_MODEL, 1.f);
    gemm_wmma<float, 0><<<dim3(32, 8, 1), blk, 0, stream>>>(
        xb, Wk_l, kf, nullptr, S_TOK, D_MODEL, D_MODEL, 0, 0, 0, D_MODEL, 1.f);
    gemm_wmma<float, 0><<<dim3(32, 8, 1), blk, 0, stream>>>(
        xb, Wv_l, vf, nullptr, S_TOK, D_MODEL, D_MODEL, 0, 0, 0, D_MODEL, 1.f);
    gemm_wmma<float, 0><<<dim3(32, 4, 1), blk, 0, stream>>>(
        memb, Wmk_l, mkf, nullptr, M_MEM, D_MODEL, D_MODEL, 0, 0, 0, D_MODEL, 1.f);
    gemm_wmma<float, 0><<<dim3(32, 4, 1), blk, 0, stream>>>(
        memb, Wmv_l, mvf, nullptr, M_MEM, D_MODEL, D_MODEL, 0, 0, 0, D_MODEL, 1.f);

    rope_q_kernel<<<(N_HEADS * S_TOK * 64) / 256, blk, 0, stream>>>(qf, Qr);
    rope_k_kernel<<<(N_HEADS * KV_LEN * 64) / 256, blk, 0, stream>>>(mkf, kf, Kr);
    vt_kernel<<<(N_HEADS * KV_LEN * 128) / 256, blk, 0, stream>>>(mvf, vf, Vt);

    gemm_wmma<unsigned short, 3><<<dim3(12, 8, 32), blk, 0, stream>>>(
        Qr, Kr, scor, nullptr, S_TOK, KV_LEN, D_HEAD,
        (long long)S_TOK * D_HEAD, (long long)KV_LEN * D_HEAD,
        (long long)S_TOK * KV_LEN, KV_LEN, scale);
    softmax_kernel<<<N_HEADS * S_TOK, blk, 0, stream>>>(scor, Pb);
    gemm_wmma<unsigned short, 2><<<dim3(1, 8, 32), blk, 0, stream>>>(
        Pb, Vt, ob, nullptr, S_TOK, D_HEAD, KV_LEN,
        (long long)S_TOK * KV_LEN, (long long)D_HEAD * KV_LEN,
        (long long)D_HEAD, D_MODEL, 1.f);
    gemm_wmma<float, 1><<<dim3(32, 8, 1), blk, 0, stream>>>(
        ob, Wo_l, hB, hB, S_TOK, D_MODEL, D_MODEL, 0, 0, 0, D_MODEL, 1.f);

    rmsnorm_kernel<<<S_TOK, blk, 0, stream>>>(hB, ln2 + l * D_MODEL, xb);
    gemm_wmma<float, 0><<<dim3(64, 8, 1), blk, 0, stream>>>(
        xb, Wg_l, gf, nullptr, S_TOK, F_FFN, D_MODEL, 0, 0, 0, F_FFN, 1.f);
    gemm_wmma<float, 0><<<dim3(64, 8, 1), blk, 0, stream>>>(
        xb, Wu_l, uf, nullptr, S_TOK, F_FFN, D_MODEL, 0, 0, 0, F_FFN, 1.f);
    silu_mul_kernel<<<(S_TOK * F_FFN) / 256, blk, 0, stream>>>(
        gf, uf, ffb, S_TOK * F_FFN);
    gemm_wmma<float, 1><<<dim3(32, 8, 1), blk, 0, stream>>>(
        ffb, Wd_l, hB, hB, S_TOK, D_MODEL, F_FFN, 0, 0, 0, D_MODEL, 1.f);
  }

  rmsnorm_kernel<<<S_TOK, blk, 0, stream>>>(hB, normw, xb);
  gemm_wmma<float, 0><<<dim3(64, 8, 1), blk, 0, stream>>>(
      xb, lmh, (float*)d_out, nullptr, S_TOK, V_VOCAB, D_MODEL, 0, 0, 0,
      V_VOCAB, 1.f);
}
